// AttentionTypeConcatSheafLearner_31842887533258
// MI455X (gfx1250) — matrix operand order repaired and checked by reading, not compiled, not run on hardware
//
#include <hip/hip_runtime.h>

typedef _Float16 v8h  __attribute__((ext_vector_type(8)));
typedef _Float16 v16h __attribute__((ext_vector_type(16)));
typedef float    v8f  __attribute__((ext_vector_type(8)));

#define C_F   128
#define H_DIM 64
#define TOT   288
#define EPSN  1e-5f
#define KT1   9     // 288 / 32 K-steps for GEMM1
#define NT1   4     // 64 / 16 N-tiles for GEMM1
#define KT2   2     // 64 / 32 K-steps for GEMM2
#define WPB   2     // waves per block

// ---------------------------------------------------------------------------
// Pack W1 [288,64] and W2 [64,16] (f32, row-major) into f16 fragments laid out
// exactly as the per-lane B-operand of v_wmma_f32_16x16x32_f16 expects:
//   lane 0-15  : n = lane,    holds K = kbase+{0..7, 16..23}  (16 halves)
//   lane 16-31 : n = lane-16, holds K = kbase+{8..15, 24..31}
// pW1[((kt*4+nt)*32 + lane)*16 + j], pW2[(kt*32 + lane)*16 + j]
// ---------------------------------------------------------------------------
__global__ void pack_weights(const float* __restrict__ W1, const float* __restrict__ W2,
                             _Float16* __restrict__ pW1, _Float16* __restrict__ pW2) {
  int t = blockIdx.x * blockDim.x + threadIdx.x;
  const int n1 = KT1 * NT1 * 32 * 16;
  if (t < n1) {
    int j = t & 15, ls = t >> 4;
    int lane = ls & 31, frag = ls >> 5;
    int nt = frag & 3, kt = frag >> 2;
    int k = kt * 32 + (j & 8) * 2 + (lane >> 4) * 8 + (j & 7);
    int n = nt * 16 + (lane & 15);
    pW1[t] = (_Float16)W1[k * H_DIM + n];
  } else {
    int t2 = t - n1;
    if (t2 < KT2 * 32 * 16) {
      int j = t2 & 15, ls = t2 >> 4;
      int lane = ls & 31, kt = ls >> 5;
      int k = kt * 32 + (j & 8) * 2 + (lane >> 4) * 8 + (j & 7);
      int n = lane & 15;
      pW2[t2] = (_Float16)W2[k * 16 + n];
    }
  }
}

// Build/normalize an 8-feature chunk and store f16 into LDS (j0 is a constant
// after full unroll, so the branches fold away).
__device__ __forceinline__ void emit_chunk(int j0, const float* xr, const float* xc,
                                           int ntr, int ntc, int et,
                                           float mean, float rs,
                                           const float* lnw, const float* lnb,
                                           _Float16* dst) {
  float v[8];
  if (j0 < 128) {
    float4 a = *(const float4*)(xr + j0);
    float4 b = *(const float4*)(xr + j0 + 4);
    v[0]=a.x; v[1]=a.y; v[2]=a.z; v[3]=a.w; v[4]=b.x; v[5]=b.y; v[6]=b.z; v[7]=b.w;
  } else if (j0 < 256) {
    const float* p = xc + (j0 - 128);
    float4 a = *(const float4*)(p);
    float4 b = *(const float4*)(p + 4);
    v[0]=a.x; v[1]=a.y; v[2]=a.z; v[3]=a.w; v[4]=b.x; v[5]=b.y; v[6]=b.z; v[7]=b.w;
  } else if (j0 < 264) {
    #pragma unroll
    for (int i = 0; i < 8; ++i) v[i] = (ntr == (j0 - 256 + i)) ? 1.f : 0.f;
  } else if (j0 < 272) {
    #pragma unroll
    for (int i = 0; i < 8; ++i) v[i] = (ntc == (j0 - 264 + i)) ? 1.f : 0.f;
  } else {
    #pragma unroll
    for (int i = 0; i < 8; ++i) v[i] = (et == (j0 - 272 + i)) ? 1.f : 0.f;
  }
  float4 w0 = *(const float4*)(lnw + j0);
  float4 w1 = *(const float4*)(lnw + j0 + 4);
  float4 c0 = *(const float4*)(lnb + j0);
  float4 c1 = *(const float4*)(lnb + j0 + 4);
  float wv[8] = {w0.x,w0.y,w0.z,w0.w,w1.x,w1.y,w1.z,w1.w};
  float bv[8] = {c0.x,c0.y,c0.z,c0.w,c1.x,c1.y,c1.z,c1.w};
  v8h h8;
  #pragma unroll
  for (int i = 0; i < 8; ++i)
    h8[i] = (_Float16)((v[i] - mean) * rs * wv[i] + bv[i]);
  *(v8h*)(dst + j0) = h8;
}

__device__ __forceinline__ v16h cat8(v8h lo, v8h hi) {
  return __builtin_shufflevector(lo, hi, 0,1,2,3,4,5,6,7,8,9,10,11,12,13,14,15);
}

// ---------------------------------------------------------------------------
// Main kernel: 1 wave == 1 tile of 16 edges.
// ---------------------------------------------------------------------------
__global__ void __launch_bounds__(32 * WPB, 1)
edge_sheaf_kernel(const float* __restrict__ x, const int* __restrict__ ei,
                  const int* __restrict__ etypes, const int* __restrict__ ntypes,
                  const float* __restrict__ lnw, const float* __restrict__ lnb,
                  const float* __restrict__ b1, const float* __restrict__ b2,
                  const _Float16* __restrict__ pW1, const _Float16* __restrict__ pW2,
                  float* __restrict__ out, int E) {
  __shared__ _Float16 sFeat[WPB][16 * TOT];    // normalized f16 features per tile
  __shared__ _Float16 sHid [WPB][16 * H_DIM];  // relu(hidden) per tile

  const int lane = threadIdx.x & 31;
  const int wave = threadIdx.x >> 5;
  const int hf   = lane >> 4;   // which half of the wave
  const int ml   = lane & 15;   // edge-in-tile (A rows) / n (B,C columns)
  const int tile = blockIdx.x * WPB + wave;

  const int edge = tile * 16 + ml;
  const int row  = ei[edge];
  const int col  = ei[E + edge];
  const int et   = etypes[edge];
  const int ntr  = ntypes[row];
  const int ntc  = ntypes[col];
  const float* xr = x + (long long)row * C_F;
  const float* xc = x + (long long)col * C_F;

  // ---- pass 1: sum / sumsq over this lane's 144 features --------------------
  float s = 0.f, s2 = 0.f;
  if (hf == 0) {
    #pragma unroll
    for (int j = 0; j < 128; j += 4) {
      float4 v = *(const float4*)(xr + j);
      s  += v.x + v.y + v.z + v.w;
      s2 += v.x*v.x + v.y*v.y + v.z*v.z + v.w*v.w;
    }
    #pragma unroll
    for (int j = 0; j < 16; j += 4) {
      float4 v = *(const float4*)(xc + j);
      s  += v.x + v.y + v.z + v.w;
      s2 += v.x*v.x + v.y*v.y + v.z*v.z + v.w*v.w;
    }
  } else {
    #pragma unroll
    for (int j = 16; j < 128; j += 4) {
      float4 v = *(const float4*)(xc + j);
      s  += v.x + v.y + v.z + v.w;
      s2 += v.x*v.x + v.y*v.y + v.z*v.z + v.w*v.w;
    }
    s += 3.f; s2 += 3.f;  // three one-hot 1.0's live in this half
  }
  s  += __shfl_xor(s, 16);
  s2 += __shfl_xor(s2, 16);
  const float mean = s * (1.f / TOT);
  const float var  = s2 * (1.f / TOT) - mean * mean;
  const float rs   = rsqrtf(var + EPSN);

  // ---- pass 2: normalize -> f16 -> LDS (A-matrix source) -------------------
  _Float16* feat = sFeat[wave];
  _Float16* dst  = feat + ml * TOT;
  if (hf == 0) {
    #pragma unroll
    for (int c8 = 0; c8 < 18; ++c8)
      emit_chunk(c8 * 8, xr, xc, ntr, ntc, et, mean, rs, lnw, lnb, dst);
  } else {
    #pragma unroll
    for (int c8 = 0; c8 < 18; ++c8)
      emit_chunk(144 + c8 * 8, xr, xc, ntr, ntc, et, mean, rs, lnw, lnb, dst);
  }
  __syncthreads();

  // ---- GEMM1: [16x288] x [288x64] via 9x4 WMMA -----------------------------
  v8f acc[NT1];
  #pragma unroll
  for (int nt = 0; nt < NT1; ++nt)
    #pragma unroll
    for (int i = 0; i < 8; ++i) acc[nt][i] = 0.f;

  const _Float16* aRow = feat + ml * TOT;  // A: row m = lane&15
  #pragma unroll
  for (int kt = 0; kt < KT1; ++kt) {
    // A fragment: lane<16 holds K {kb..kb+7, kb+16..kb+23}; lane>=16 offset +8
    const int kb = kt * 32 + hf * 8;
    v8h alo = *(const v8h*)(aRow + kb);
    v8h ahi = *(const v8h*)(aRow + kb + 16);
    v16h A = cat8(alo, ahi);
    const _Float16* bp = pW1 + ((kt * NT1) * 32 + lane) * 16;
    #pragma unroll
    for (int nt = 0; nt < NT1; ++nt) {
      v16h B = *(const v16h*)(bp + nt * (32 * 16));
      acc[nt] = __builtin_amdgcn_wmma_f32_16x16x32_f16(
          false, A, false, B, (short)0, acc[nt], false, false);
    }
  }

  // ---- bias + ReLU -> f16 hidden in LDS (re-layout for GEMM2 A) ------------
  _Float16* hid = sHid[wave];
  #pragma unroll
  for (int nt = 0; nt < NT1; ++nt) {
    const float bias = b1[nt * 16 + ml];       // C layout: n = lane&15
    #pragma unroll
    for (int r = 0; r < 8; ++r) {              // C layout: m = r + 8*hf
      float hv = acc[nt][r] + bias;
      hv = hv > 0.f ? hv : 0.f;
      hid[(r + 8 * hf) * H_DIM + nt * 16 + ml] = (_Float16)hv;
    }
  }
  __syncthreads();

  // ---- GEMM2: [16x64] x [64x16] via 2 WMMA ---------------------------------
  v8f acc2;
  #pragma unroll
  for (int i = 0; i < 8; ++i) acc2[i] = 0.f;
  const _Float16* hRow = hid + ml * H_DIM;
  #pragma unroll
  for (int kt = 0; kt < KT2; ++kt) {
    const int kb = kt * 32 + hf * 8;
    v8h alo = *(const v8h*)(hRow + kb);
    v8h ahi = *(const v8h*)(hRow + kb + 16);
    v16h A = cat8(alo, ahi);
    v16h B = *(const v16h*)(pW2 + (kt * 32 + lane) * 16);
    acc2 = __builtin_amdgcn_wmma_f32_16x16x32_f16(
        false, A, false, B, (short)0, acc2, false, false);
  }

  // ---- bias2, 4-wide row softmax (quad shuffles), I - att, store -----------
  const float bias2 = b2[ml];
  const int   rout = ml >> 2, cout = ml & 3;
  const float diag = (rout == cout) ? 1.f : 0.f;
  #pragma unroll
  for (int r = 0; r < 8; ++r) {
    float v  = acc2[r] + bias2;
    float mx = fmaxf(v, __shfl_xor(v, 1));
    mx       = fmaxf(mx, __shfl_xor(mx, 2));
    float e  = __expf(v - mx);
    float se = e + __shfl_xor(e, 1);
    se       = se + __shfl_xor(se, 2);
    const int m = r + 8 * hf;
    out[(long long)(tile * 16 + m) * 16 + ml] = diag - e / se;
  }
}

extern "C" void kernel_launch(void* const* d_in, const int* in_sizes, int n_in,
                              void* d_out, int out_size, void* d_ws, size_t ws_size,
                              hipStream_t stream) {
  const float* x      = (const float*)d_in[0];
  const int*   ei     = (const int*)d_in[1];
  const int*   etypes = (const int*)d_in[2];
  const int*   ntypes = (const int*)d_in[3];
  const float* lnw    = (const float*)d_in[4];
  const float* lnb    = (const float*)d_in[5];
  const float* W1     = (const float*)d_in[6];
  const float* b1     = (const float*)d_in[7];
  const float* W2     = (const float*)d_in[8];
  const float* b2     = (const float*)d_in[9];
  float* out = (float*)d_out;
  const int E = in_sizes[2];  // edge_types count

  _Float16* pW1 = (_Float16*)d_ws;
  _Float16* pW2 = pW1 + KT1 * NT1 * 32 * 16;

  const int packTotal = KT1 * NT1 * 32 * 16 + KT2 * 32 * 16;
  pack_weights<<<(packTotal + 255) / 256, 256, 0, stream>>>(W1, W2, pW1, pW2);

  const int ntiles = E / 16;                       // 50000, E % 16 == 0
  const int blocks = (ntiles + WPB - 1) / WPB;     // 25000, exact
  edge_sheaf_kernel<<<blocks, 32 * WPB, 0, stream>>>(
      x, ei, etypes, ntypes, lnw, lnb, b1, b2, pW1, pW2, out, E);
}
